// CustomTransformerLayer_22402549416298
// MI455X (gfx1250) — compile-verified, matrix operand order checked by reading, not run on hardware
//
#include <hip/hip_runtime.h>
#include <hip/hip_bf16.h>
#include <math.h>

// ---------------------------------------------------------------------------
// Transformer layer for MI455X (gfx1250): bf16 WMMA GEMMs + fused attention.
// B=8, S=1024, D=1024, H=16, HD=64.
// GEMM tiles staged with GLOBAL_LOAD_ASYNC_TO_LDS_B128 + double-buffered LDS.
// ---------------------------------------------------------------------------

#define B_  8
#define S_  1024
#define D_  1024
#define H_  16
#define HD_ 64

typedef __bf16 bf16_t;
typedef __attribute__((ext_vector_type(16))) __bf16 v16bf;
typedef __attribute__((ext_vector_type(8)))  __bf16 v8bf;
typedef __attribute__((ext_vector_type(8)))  float  v8f;

union BF16x16 { v16bf v; v8bf h[2]; };

__device__ inline v16bf make_frag(v8bf lo, v8bf hi) {
    BF16x16 u; u.h[0] = lo; u.h[1] = hi; return u.v;
}

__device__ inline v8f zero8() {
    v8f z; for (int e = 0; e < 8; ++e) z[e] = 0.0f; return z;
}

__device__ inline v8f wmma_bf16(v16bf a, v16bf b, v8f c) {
    // D = A(16x32 bf16) * B(32x16 bf16) + C(16x16 f32)
    return __builtin_amdgcn_wmma_f32_16x16x32_bf16(
        false, a, false, b, (short)0, c, false, false);
}

// Per ISA 10.2: generic->LDS mapping uses addr[31:0] as the wave-relative LDS
// address, so truncating a generic pointer to a __shared__ object gives the
// VDST operand for async-to-LDS instructions.
__device__ inline unsigned lds_addr32(const void* p) {
    return (unsigned)(size_t)p;
}

// CDNA5 async global->LDS copy (ASYNCcnt-tracked, no VGPR data movement).
__device__ inline void async_g2l_b128(unsigned lds_off, const void* gptr) {
    asm volatile("global_load_async_to_lds_b128 %0, %1, off"
                 :: "v"(lds_off), "v"(gptr)
                 : "memory");
}
__device__ inline void wait_async0() {
    asm volatile("s_wait_asynccnt 0" ::: "memory");
}

// Load a 16x32 (or 32x16 col-view) fragment from K-major storage:
// lanes 0-15: elems 0..7 = K+0..7, elems 8..15 = K+16..23
// lanes 16-31: elems 0..7 = K+8..15, elems 8..15 = K+24..31
__device__ inline v16bf load_frag_nt(const bf16_t* __restrict__ p, int ld,
                                     int row0, int k0, int lane) {
    int r  = row0 + (lane & 15);
    int kh = k0 + ((lane >> 4) << 3);
    const bf16_t* q = p + (size_t)r * ld + kh;
    v8bf lo = *(const v8bf*)q;
    v8bf hi = *(const v8bf*)(q + 16);
    return make_frag(lo, hi);
}

// ---------------------------------------------------------------------------
// fp32 -> bf16 conversion (grid-stride)
// ---------------------------------------------------------------------------
__global__ __launch_bounds__(256) void cvt_f32_bf16(const float* __restrict__ in,
                                                    bf16_t* __restrict__ out,
                                                    size_t n) {
    size_t i = (size_t)blockIdx.x * blockDim.x + threadIdx.x;
    size_t stride = (size_t)gridDim.x * blockDim.x;
    for (; i < n; i += stride) out[i] = (bf16_t)in[i];
}

// ---------------------------------------------------------------------------
// NT GEMM: C[M,N] = A[M,K] * B[N,K]^T (+bias, optional ReLU)
// block tile 128x64, K-step 32, 8 waves, each wave 32x32 (2x2 wmma tiles)
// Tiles staged via async global->LDS copies, double buffered.
// ---------------------------------------------------------------------------
template<bool RELU, bool OUTF, bool OUTB>
__global__ __launch_bounds__(256) void gemm_nt_bf16(
    const bf16_t* __restrict__ A, const bf16_t* __restrict__ Bm,
    const float* __restrict__ bias,
    float* __restrict__ C, bf16_t* __restrict__ Cb,
    int M, int N, int K)
{
    __shared__ __align__(16) bf16_t sA[2][128][40];
    __shared__ __align__(16) bf16_t sB[2][64][40];

    const int t    = threadIdx.x;
    const int lane = t & 31;
    const int w    = t >> 5;
    const int wm   = w >> 1;     // 0..3  (M direction)
    const int wn   = w & 1;      // 0..1  (N direction)
    const int m0   = blockIdx.y * 128;
    const int n0   = blockIdx.x * 64;

    v8f c[2][2];
    c[0][0] = zero8(); c[0][1] = zero8(); c[1][0] = zero8(); c[1][1] = zero8();

    const int arow = t >> 1, ahalf = t & 1;   // 2 threads / A row, 16 elems each
    const int brow = t >> 2, bq    = t & 3;   // 4 threads / B row, 8 elems each

    // Issue async copies for the K-tile starting at k0 into LDS buffer `buf`.
    auto issue_tile = [&](int k0, int buf) {
        const bf16_t* srca = A  + (size_t)(m0 + arow) * K + k0 + ahalf * 16;
        const bf16_t* srcb = Bm + (size_t)(n0 + brow) * K + k0 + bq * 8;
        async_g2l_b128(lds_addr32(&sA[buf][arow][ahalf * 16]),     srca);
        async_g2l_b128(lds_addr32(&sA[buf][arow][ahalf * 16 + 8]), srca + 8);
        async_g2l_b128(lds_addr32(&sB[buf][brow][bq * 8]),         srcb);
        if (k0 + 64 < K) {                    // L2 prefetch for tile after next
            __builtin_prefetch(srca + 64, 0, 0);
            __builtin_prefetch(srcb + 64, 0, 0);
        }
    };

    const int nk = K >> 5;
    issue_tile(0, 0);

    for (int kt = 0; kt < nk; ++kt) {
        const int cur = kt & 1;
        wait_async0();        // this wave's copies into buf `cur` complete
        __syncthreads();      // whole tile visible; prev buffer fully consumed
        if (kt + 1 < nk) issue_tile((kt + 1) * 32, cur ^ 1);

        v16bf a0 = load_frag_nt(&sA[cur][0][0], 40, wm * 32,      0, lane);
        v16bf a1 = load_frag_nt(&sA[cur][0][0], 40, wm * 32 + 16, 0, lane);
        v16bf b0 = load_frag_nt(&sB[cur][0][0], 40, wn * 32,      0, lane);
        v16bf b1 = load_frag_nt(&sB[cur][0][0], 40, wn * 32 + 16, 0, lane);

        c[0][0] = wmma_bf16(a0, b0, c[0][0]);
        c[0][1] = wmma_bf16(a0, b1, c[0][1]);
        c[1][0] = wmma_bf16(a1, b0, c[1][0]);
        c[1][1] = wmma_bf16(a1, b1, c[1][1]);
    }

    const int mbase = m0 + wm * 32;
    const int nbase = n0 + wn * 32;
    for (int i = 0; i < 2; ++i) {
        for (int j = 0; j < 2; ++j) {
            int nn = nbase + j * 16 + (lane & 15);
            float bv = bias ? bias[nn] : 0.0f;
            for (int r = 0; r < 8; ++r) {
                int mm = mbase + i * 16 + r + ((lane >> 4) << 3);
                float val = c[i][j][r] + bv;
                if (RELU) val = fmaxf(val, 0.0f);
                if (OUTF) C [(size_t)mm * N + nn] = val;
                if (OUTB) Cb[(size_t)mm * N + nn] = (bf16_t)val;
            }
        }
    }
}

// ---------------------------------------------------------------------------
// Fused attention: one block = (b, h, 32 q-rows).
// phase1: scores = q*k^T (WMMA) -> LDS fp32 [32][1024]
// phase2: row max / sum (wave32 shuffles)
// phase3: fp32 probs -> d_out
// phase4: ctx = softmax(scores) * v (WMMA, A rebuilt bf16 from LDS)
// Dynamic LDS: 32*1024*4 + 64*40*2 + 64*4 = 136448 bytes (<320KB/WGP).
// ---------------------------------------------------------------------------
#define ATT_LDS_BYTES (32 * 1024 * 4 + 64 * 40 * 2 + 64 * 4)

__global__ __launch_bounds__(256) void attn_kernel(
    const bf16_t* __restrict__ qkv,   // [B,S,3D] bf16 (q|k|v)
    float* __restrict__ probs_out,    // [B,H,S,S] fp32
    bf16_t* __restrict__ ctx_out)     // [B,S,D]  bf16
{
    extern __shared__ __align__(16) char smem_raw[];
    float*  sS    = (float*)smem_raw;                          // [32][1024]
    bf16_t* sVT   = (bf16_t*)(smem_raw + 32 * 1024 * 4);       // [64][40] v^T
    float*  sStat = (float*)(smem_raw + 32 * 1024 * 4 + 64 * 40 * 2); // [64]

    const int t    = threadIdx.x;
    const int lane = t & 31;
    const int w    = t >> 5;
    const int q0   = blockIdx.x * 32;
    const int h    = blockIdx.y;
    const int b    = blockIdx.z;
    const float scale = 0.125f;   // 1/sqrt(64)

    const size_t row_stride = (size_t)3 * D_;
    const bf16_t* qkv_b = qkv + (size_t)b * S_ * 3 * D_;

    // ---- phase 1: scores (unscaled) ----
    v16bf aq[2][2];
    for (int i = 0; i < 2; ++i)
        for (int kc = 0; kc < 2; ++kc) {
            int row = q0 + i * 16 + (lane & 15);
            int kh  = kc * 32 + ((lane >> 4) << 3);
            const bf16_t* p = qkv_b + (size_t)row * row_stride + h * HD_ + kh;
            aq[i][kc] = make_frag(*(const v8bf*)p, *(const v8bf*)(p + 16));
        }

    for (int nt = 0; nt < 8; ++nt) {
        int ncol0 = w * 128 + nt * 16;
        v16bf bk[2];
        for (int kc = 0; kc < 2; ++kc) {
            int key = ncol0 + (lane & 15);
            int kh  = kc * 32 + ((lane >> 4) << 3);
            const bf16_t* p = qkv_b + (size_t)key * row_stride + D_ + h * HD_ + kh;
            bk[kc] = make_frag(*(const v8bf*)p, *(const v8bf*)(p + 16));
        }
        v8f c0 = zero8(), c1 = zero8();
        c0 = wmma_bf16(aq[0][0], bk[0], c0);
        c0 = wmma_bf16(aq[0][1], bk[1], c0);
        c1 = wmma_bf16(aq[1][0], bk[0], c1);
        c1 = wmma_bf16(aq[1][1], bk[1], c1);
        int col = ncol0 + (lane & 15);
        int mh  = (lane >> 4) << 3;
        for (int r = 0; r < 8; ++r) {
            sS[(r + mh) * 1024 + col]      = c0[r];
            sS[(16 + r + mh) * 1024 + col] = c1[r];
        }
    }
    __syncthreads();

    // ---- phase 2: per-row max & sum of exp ----
    {
        int row = t >> 3;          // 0..31
        int seg = t & 7;           // 8 lanes per row (same wave)
        float m = -3.0e38f;
        for (int cc = 0; cc < 128; ++cc) {
            float v = sS[row * 1024 + seg * 128 + cc] * scale;
            m = fmaxf(m, v);
        }
        m = fmaxf(m, __shfl_xor(m, 1, 32));
        m = fmaxf(m, __shfl_xor(m, 2, 32));
        m = fmaxf(m, __shfl_xor(m, 4, 32));
        float s = 0.0f;
        for (int cc = 0; cc < 128; ++cc) {
            float v = sS[row * 1024 + seg * 128 + cc] * scale;
            s += expf(v - m);
        }
        s += __shfl_xor(s, 1, 32);
        s += __shfl_xor(s, 2, 32);
        s += __shfl_xor(s, 4, 32);
        if (seg == 0) { sStat[row] = m; sStat[32 + row] = s; }
    }
    __syncthreads();

    // ---- phase 3: stream fp32 probs to d_out ----
    {
        float* pout = probs_out + ((size_t)(b * H_ + h) * S_ + q0) * S_;
        for (int it = 0; it < 128; ++it) {
            int e   = t + it * 256;
            int row = e >> 10;
            int col = e & 1023;
            float m   = sStat[row];
            float inv = 1.0f / sStat[32 + row];
            pout[(size_t)row * S_ + col] = expf(sS[e] * scale - m) * inv;
        }
    }

    // ---- phase 4: ctx = probs * v ----
    const int mtile = w >> 2;     // 0..1
    const int ntile = w & 3;      // 0..3
    v8f cc = zero8();

    for (int kc = 0; kc < 32; ++kc) {
        int kbase = kc * 32;
        __syncthreads();   // previous v^T tile fully consumed
        {
            int keyr = t >> 3;          // 0..31
            int hd0  = (t & 7) * 8;     // 0..56
            const bf16_t* vp = qkv_b + (size_t)(kbase + keyr) * row_stride
                                     + 2 * D_ + h * HD_ + hd0;
            v8bf vv = *(const v8bf*)vp;
            for (int i = 0; i < 8; ++i) sVT[(hd0 + i) * 40 + keyr] = vv[i];
        }
        __syncthreads();

        // A fragment: bf16 probs rebuilt from fp32 scores in LDS
        int arow = mtile * 16 + (lane & 15);
        int kh   = (lane >> 4) << 3;
        float m   = sStat[arow];
        float inv = 1.0f / sStat[32 + arow];
        v16bf af;
        for (int e = 0; e < 8; ++e) {
            float s0 = sS[arow * 1024 + kbase + kh + e];
            float s1 = sS[arow * 1024 + kbase + kh + 16 + e];
            af[e]     = (bf16_t)(expf(s0 * scale - m) * inv);
            af[e + 8] = (bf16_t)(expf(s1 * scale - m) * inv);
        }
        // B fragment: v^T (column = hd, contiguous K in LDS)
        int ncol = ntile * 16 + (lane & 15);
        v8bf lo = *(const v8bf*)&sVT[ncol * 40 + kh];
        v8bf hi = *(const v8bf*)&sVT[ncol * 40 + kh + 16];
        v16bf bfm = make_frag(lo, hi);

        cc = wmma_bf16(af, bfm, cc);
    }

    int mh = (lane >> 4) << 3;
    int nc = h * HD_ + ntile * 16 + (lane & 15);
    for (int r = 0; r < 8; ++r) {
        int mrow = q0 + mtile * 16 + r + mh;
        ctx_out[(size_t)(b * S_ + mrow) * D_ + nc] = (bf16_t)cc[r];
    }
}

// ---------------------------------------------------------------------------
// Residual + LayerNorm: out = LN(X + R) * g + b   (one row / block, D=1024)
// ---------------------------------------------------------------------------
template<bool OUTB>
__global__ __launch_bounds__(256) void ln_residual(
    const float* __restrict__ X, const float* __restrict__ R,
    const float* __restrict__ g, const float* __restrict__ bta,
    float* __restrict__ outf, bf16_t* __restrict__ outb)
{
    __shared__ float red[16];
    const int row = blockIdx.x;
    const int t   = threadIdx.x;
    const size_t base = (size_t)row * D_;

    float vals[4];
    float s = 0.0f, s2 = 0.0f;
    for (int i = 0; i < 4; ++i) {
        int c = t + i * 256;
        float v = X[base + c] + R[base + c];
        vals[i] = v; s += v; s2 += v * v;
    }
    for (int m = 16; m >= 1; m >>= 1) {
        s  += __shfl_xor(s,  m, 32);
        s2 += __shfl_xor(s2, m, 32);
    }
    const int w = t >> 5, lane = t & 31;
    if (lane == 0) { red[w] = s; red[8 + w] = s2; }
    __syncthreads();
    if (t == 0) {
        float a = 0.0f, b2 = 0.0f;
        for (int i = 0; i < 8; ++i) { a += red[i]; b2 += red[8 + i]; }
        red[0] = a; red[8] = b2;
    }
    __syncthreads();
    float mean = red[0] * (1.0f / D_);
    float var  = red[8] * (1.0f / D_) - mean * mean;
    float rstd = rsqrtf(var + 1e-5f);
    for (int i = 0; i < 4; ++i) {
        int c = t + i * 256;
        float o = (vals[i] - mean) * rstd * g[c] + bta[c];
        outf[base + c] = o;
        if (OUTB) outb[base + c] = (bf16_t)o;
    }
}

// ---------------------------------------------------------------------------
// Host launcher
// ---------------------------------------------------------------------------
extern "C" void kernel_launch(void* const* d_in, const int* in_sizes, int n_in,
                              void* d_out, int out_size, void* d_ws, size_t ws_size,
                              hipStream_t stream)
{
    (void)in_sizes; (void)n_in; (void)out_size; (void)ws_size;

    const float* x    = (const float*)d_in[0];
    const float* Wqkv = (const float*)d_in[1];
    const float* bqkv = (const float*)d_in[2];
    const float* Wout = (const float*)d_in[3];
    const float* bout = (const float*)d_in[4];
    const float* W1   = (const float*)d_in[5];
    const float* b1   = (const float*)d_in[6];
    const float* W2   = (const float*)d_in[7];
    const float* b2   = (const float*)d_in[8];
    const float* ln1g = (const float*)d_in[9];
    const float* ln1b = (const float*)d_in[10];
    const float* ln2g = (const float*)d_in[11];
    const float* ln2b = (const float*)d_in[12];

    float* x2_out    = (float*)d_out;                       // [B,S,D]
    float* probs_out = x2_out + (size_t)B_ * S_ * D_;       // [B,H,S,S]

    // workspace carve-out
    char* ws = (char*)d_ws;
    size_t off = 0;
    auto alloc = [&](size_t bytes) -> char* {
        char* p = ws + off;
        off += (bytes + 255) & ~(size_t)255;
        return p;
    };
    const size_t NX = (size_t)B_ * S_ * D_;                 // 8388608
    bf16_t* xb    = (bf16_t*)alloc(NX * 2);
    bf16_t* Wqkvb = (bf16_t*)alloc((size_t)3 * D_ * D_ * 2);
    bf16_t* Woutb = (bf16_t*)alloc((size_t)D_ * D_ * 2);
    bf16_t* W1b   = (bf16_t*)alloc((size_t)2 * D_ * D_ * 2);
    bf16_t* W2b   = (bf16_t*)alloc((size_t)2 * D_ * D_ * 2);
    bf16_t* qkvb  = (bf16_t*)alloc((size_t)B_ * S_ * 3 * D_ * 2);
    bf16_t* ctxb  = (bf16_t*)alloc(NX * 2);
    float*  attn_out = (float*)alloc(NX * 4);
    float*  x1    = (float*)alloc(NX * 4);
    bf16_t* x1b   = (bf16_t*)alloc(NX * 2);
    bf16_t* hb    = (bf16_t*)alloc((size_t)B_ * S_ * 2 * D_ * 2);
    float*  ff    = (float*)alloc(NX * 4);

    // 1) bf16 conversions
    cvt_f32_bf16<<<dim3(2048), 256, 0, stream>>>(x,    xb,    NX);
    cvt_f32_bf16<<<dim3(1024), 256, 0, stream>>>(Wqkv, Wqkvb, (size_t)3 * D_ * D_);
    cvt_f32_bf16<<<dim3(512),  256, 0, stream>>>(Wout, Woutb, (size_t)D_ * D_);
    cvt_f32_bf16<<<dim3(512),  256, 0, stream>>>(W1,   W1b,   (size_t)2 * D_ * D_);
    cvt_f32_bf16<<<dim3(512),  256, 0, stream>>>(W2,   W2b,   (size_t)2 * D_ * D_);

    const int M = B_ * S_;   // 8192

    // 2) qkv = x @ Wqkv^T + bqkv   (bf16 out)
    gemm_nt_bf16<false, false, true><<<dim3(3 * D_ / 64, M / 128), 256, 0, stream>>>(
        xb, Wqkvb, bqkv, nullptr, qkvb, M, 3 * D_, D_);

    // 3) fused attention: scores -> softmax -> probs(d_out) + ctx(bf16)
    hipFuncSetAttribute(reinterpret_cast<const void*>(attn_kernel),
                        hipFuncAttributeMaxDynamicSharedMemorySize, ATT_LDS_BYTES);
    attn_kernel<<<dim3(S_ / 32, H_, B_), 256, ATT_LDS_BYTES, stream>>>(
        qkvb, probs_out, ctxb);

    // 4) attn_out = ctx @ Wout^T + bout  (fp32 out)
    gemm_nt_bf16<false, true, false><<<dim3(D_ / 64, M / 128), 256, 0, stream>>>(
        ctxb, Woutb, bout, attn_out, nullptr, M, D_, D_);

    // 5) x1 = LN(x + attn_out)  (fp32 + bf16 copy)
    ln_residual<true><<<dim3(M), 256, 0, stream>>>(x, attn_out, ln1g, ln1b, x1, x1b);

    // 6) h = relu(x1 @ W1^T + b1)  (bf16 out)
    gemm_nt_bf16<true, false, true><<<dim3(2 * D_ / 64, M / 128), 256, 0, stream>>>(
        x1b, W1b, b1, nullptr, hb, M, 2 * D_, D_);

    // 7) ff = h @ W2^T + b2  (fp32 out)
    gemm_nt_bf16<false, true, false><<<dim3(D_ / 64, M / 128), 256, 0, stream>>>(
        hb, W2b, b2, ff, nullptr, M, D_, 2 * D_);

    // 8) x2 = LN(x1 + ff) -> d_out
    ln_residual<false><<<dim3(M), 256, 0, stream>>>(x1, ff, ln2g, ln2b, x2_out, nullptr);
}
